// Xerxes2DecoderLayer_52673478918596
// MI455X (gfx1250) — compile-verified, matrix operand order checked by reading, not run on hardware
//
#include <hip/hip_runtime.h>
#include <hip/hip_bf16.h>
#include <stdint.h>

typedef __attribute__((ext_vector_type(16))) __bf16 v16bf;
typedef __attribute__((ext_vector_type(8)))  float  v8f;

#define S_LEN  2048
#define HIDDIM 2048
#define NHEAD  16
#define NOPE_D 128
#define ROPE_D 64
#define VDIM   128
#define QHD    192   // NOPE+ROPE
#define QLAT   1536
#define KVLAT  512
#define INTERD 8192

// ---------------- helpers ----------------
__device__ __forceinline__ __bf16 f2bf(float f) {
    union { float f; unsigned u; } x; x.f = f;
    unsigned r = x.u + 0x7FFFu + ((x.u >> 16) & 1u);
    return __builtin_bit_cast(__bf16, (unsigned short)(r >> 16));
}

__device__ __forceinline__ v8f zero8() {
    v8f z;
#pragma unroll
    for (int i = 0; i < 8; ++i) z[i] = 0.0f;
    return z;
}

// ISA 7.12.2 bf16 A/B fragment K mapping; kh = (lane>>4)*8.
__device__ __forceinline__ int kmap(int v, int kh) {
    return ((v & 4) << 2) + kh + 2 * (v & 3);
}

// One lane's slice of a 16x32 bf16 fragment; v=0..3 cover 8 contiguous bf16
// (merge to ds_load_b128), v=4..7 the other 16B.
__device__ __forceinline__ v16bf load_frag(const __bf16* rowptr, int k0, int kh) {
    v16bf f;
#pragma unroll
    for (int v = 0; v < 8; ++v) {
        int k = k0 + kmap(v, kh);
        unsigned u = *(const unsigned*)(rowptr + k);
        f[2 * v]     = __builtin_bit_cast(__bf16, (unsigned short)(u & 0xffffu));
        f[2 * v + 1] = __builtin_bit_cast(__bf16, (unsigned short)(u >> 16));
    }
    return f;
}

__device__ __forceinline__ float block_sum_256(float v) {
#pragma unroll
    for (int off = 16; off > 0; off >>= 1) v += __shfl_xor(v, off, 32);
    __shared__ float red[8];
    int lane = threadIdx.x & 31, wv = threadIdx.x >> 5;
    if (lane == 0) red[wv] = v;
    __syncthreads();
    float t = 0.0f;
#pragma unroll
    for (int i = 0; i < 8; ++i) t += red[i];
    __syncthreads();
    return t;
}

// ---------------- bf16 WMMA GEMM: C(f32, MxN) = A(MxK) * B(KxN) ----------------
// 128 threads = 4 waves; block tile 128x128; wave tile 64x64 (16 WMMAs/K-step).
// Register-staged LDS double buffering: next tile's global loads issue before
// the WMMA block so HBM latency hides under matrix math. M%128==0, K%32==0.
#define BM 128
#define BN 128
#define BK 32
#define LDT 40   // padded LDS stride (bf16), 80B rows keep 16B alignment

template <bool RAGGED_N>
__global__ __launch_bounds__(128) void gemm_bf16_wmma(
    const __bf16* __restrict__ A, const __bf16* __restrict__ B,
    float* __restrict__ C, int M, int N, int K) {
    __shared__ __bf16 As[2][BM * LDT];
    __shared__ __bf16 Bs[2][BN * LDT];   // transposed: [n][k]
    const int tid   = threadIdx.x;
    const int lane  = tid & 31;
    const int wave  = tid >> 5;
    const int waveM = wave >> 1;                 // 0..1
    const int waveN = wave & 1;                  // 0..1
    const int rowB  = blockIdx.y * BM;
    const int colB  = blockIdx.x * BN;
    const int kh    = (lane >> 4) << 3;
    const int mm    = lane & 15;
    const int hi    = lane >> 4;

    v8f acc[4][4];
#pragma unroll
    for (int i = 0; i < 4; ++i)
#pragma unroll
        for (int j = 0; j < 4; ++j) acc[i][j] = zero8();

    __bf16 rga[32];   // A staging regs: thread t owns row t, 32 K-cols
    __bf16 rgb[32];   // B staging regs: thread t owns k=t>>2, n0=(t&3)*32

    const int bk = tid >> 2;
    const int bn0 = (tid & 3) * 32;

    auto load_tile = [&](int k0) {
        const __bf16* asrc = A + (size_t)(rowB + tid) * K + k0;
#pragma unroll
        for (int i = 0; i < 4; ++i) *(uint4*)&rga[i * 8] = *(const uint4*)(asrc + i * 8);
        if (k0 + BK < K) __builtin_prefetch(asrc + BK, 0, 1);
        const __bf16* bsrc = B + (size_t)(k0 + bk) * N + colB + bn0;
        if (!RAGGED_N || colB + bn0 + 32 <= N) {
#pragma unroll
            for (int i = 0; i < 4; ++i) *(uint4*)&rgb[i * 8] = *(const uint4*)(bsrc + i * 8);
        } else {
#pragma unroll
            for (int e = 0; e < 32; ++e)
                rgb[e] = (colB + bn0 + e < N) ? bsrc[e]
                                              : __builtin_bit_cast(__bf16, (unsigned short)0);
        }
    };
    auto store_tile = [&](int buf) {
        __bf16* adst = &As[buf][tid * LDT];
#pragma unroll
        for (int i = 0; i < 4; ++i) *(uint4*)(adst + i * 8) = *(uint4*)&rga[i * 8];
#pragma unroll
        for (int e = 0; e < 32; ++e) Bs[buf][(bn0 + e) * LDT + bk] = rgb[e];
    };
    auto compute = [&](int buf) {
        v16bf af[4], bf[4];
#pragma unroll
        for (int ms = 0; ms < 4; ++ms)
            af[ms] = load_frag(&As[buf][(waveM * 64 + ms * 16 + mm) * LDT], 0, kh);
#pragma unroll
        for (int ns = 0; ns < 4; ++ns)
            bf[ns] = load_frag(&Bs[buf][(waveN * 64 + ns * 16 + mm) * LDT], 0, kh);
#pragma unroll
        for (int ms = 0; ms < 4; ++ms)
#pragma unroll
            for (int ns = 0; ns < 4; ++ns)
                acc[ms][ns] = __builtin_amdgcn_wmma_f32_16x16x32_bf16(
                    false, af[ms], false, bf[ns], (short)0, acc[ms][ns], false, false);
    };

    load_tile(0);
    store_tile(0);
    __syncthreads();
    int cur = 0;
    for (int k0 = BK; k0 < K; k0 += BK) {
        load_tile(k0);        // global loads in flight while we do WMMAs below
        compute(cur);
        store_tile(cur ^ 1);
        __syncthreads();
        cur ^= 1;
    }
    compute(cur);

#pragma unroll
    for (int ms = 0; ms < 4; ++ms)
#pragma unroll
        for (int ns = 0; ns < 4; ++ns) {
            int col = colB + waveN * 64 + ns * 16 + mm;
            if (!RAGGED_N || col < N) {
#pragma unroll
                for (int g = 0; g < 8; ++g) {
                    int row = rowB + waveM * 64 + ms * 16 + g + 8 * hi;
                    C[(size_t)row * N + col] = acc[ms][ns][g];
                }
            }
        }
}

// ---------------- flash attention (MQA: K/V shared across 16 heads) ----------------
// grid (S/64, H), block 128 (4 waves); each wave owns 16 query rows.
__global__ __launch_bounds__(128) void flash_attn_wmma(
    const __bf16* __restrict__ qf,   // S x (H*192)
    const __bf16* __restrict__ kf,   // S x 192
    const __bf16* __restrict__ vv,   // S x 128
    __bf16* __restrict__ outb,       // S x (H*128)
    int S) {
    __shared__ __bf16 Ks[32 * 192];      // [key][d]  (async-staged)
    __shared__ __bf16 Vt[128 * LDT];     // transposed [vd][key]
    __shared__ __bf16 Ps[4][16 * LDT];   // per-wave P scratch
    const int tid  = threadIdx.x;
    const int lane = tid & 31;
    const int wave = tid >> 5;
    const int h    = blockIdx.y;
    const int qB   = blockIdx.x * 64 + wave * 16;
    const int kh   = (lane >> 4) << 3;
    const int mm   = lane & 15;
    const int hi   = lane >> 4;
    const float scale = 0.07216878364870323f;  // 192^-0.5

    // Q fragments held in registers for whole kernel
    v16bf qfr[6];
    {
        const __bf16* qrow = qf + (size_t)(qB + mm) * (NHEAD * QHD) + h * QHD;
#pragma unroll
        for (int kk = 0; kk < 6; ++kk) qfr[kk] = load_frag(qrow, kk * 32, kh);
    }

    float rmax[8], rsum[8];
    v8f accO[8];
#pragma unroll
    for (int g = 0; g < 8; ++g) { rmax[g] = -3.0e38f; rsum[g] = 0.0f; }
#pragma unroll
    for (int nt = 0; nt < 8; ++nt) accO[nt] = zero8();

    const int kvEnd = blockIdx.x * 64 + 64;
    for (int j = 0; j < kvEnd; j += 32) {
        // ---- K tile 32x192: ASYNCcnt-tracked direct global->LDS (no VGPR data) ----
#pragma unroll
        for (int i = 0; i < 6; ++i) {
            int f = tid + i * 128;           // 768 * 8 = 6144 elems
            int r = f / 24;
            int c = (f % 24) * 8;
            unsigned ldsaddr = (unsigned)(size_t)&Ks[r * 192 + c];
            const __bf16* gsrc = kf + (size_t)(j + r) * 192 + c;
            asm volatile("global_load_async_to_lds_b128 %0, %1, off"
                         :: "v"(ldsaddr), "v"(gsrc) : "memory");
        }
        // ---- cooperative V tile 32x128 -> Vt[vd][key] (transpose scatter) ----
#pragma unroll
        for (int i = 0; i < 4; ++i) {
            int f = tid + i * 128;
            int r = f >> 4;
            int c = (f & 15) * 8;
            __bf16 tmp[8];
            *(uint4*)tmp = *(const uint4*)(vv + (size_t)(j + r) * 128 + c);
#pragma unroll
            for (int e = 0; e < 8; ++e) Vt[(c + e) * LDT + r] = tmp[e];
        }
        asm volatile("s_wait_asynccnt 0x0" ::: "memory");
        __syncthreads();

        // ---- scores: 2 x (16x16) tiles over 192-dim (6 WMMAs each) ----
        v8f s2[2];
        s2[0] = zero8(); s2[1] = zero8();
#pragma unroll
        for (int nh2 = 0; nh2 < 2; ++nh2)
#pragma unroll
            for (int kk = 0; kk < 6; ++kk) {
                v16bf bfr = load_frag(&Ks[(nh2 * 16 + mm) * 192], kk * 32, kh);
                s2[nh2] = __builtin_amdgcn_wmma_f32_16x16x32_bf16(
                    false, qfr[kk], false, bfr, (short)0, s2[nh2], false, false);
            }
        // ---- scale + causal mask ----
#pragma unroll
        for (int g = 0; g < 8; ++g) {
            int row = qB + g + 8 * hi;
#pragma unroll
            for (int nh2 = 0; nh2 < 2; ++nh2) {
                int col = j + nh2 * 16 + mm;
                float v = s2[nh2][g] * scale;
                s2[nh2][g] = (col <= row) ? v : -3.0e38f;
            }
        }
        // ---- online softmax (row reductions over 16-lane halves) ----
        float fg[8];
#pragma unroll
        for (int g = 0; g < 8; ++g) {
            float t = fmaxf(s2[0][g], s2[1][g]);
#pragma unroll
            for (int off = 1; off < 16; off <<= 1) t = fmaxf(t, __shfl_xor(t, off, 32));
            float mnew = fmaxf(rmax[g], t);
            fg[g]  = __expf(rmax[g] - mnew);
            rmax[g] = mnew;
            float p0 = __expf(s2[0][g] - mnew);
            float p1 = __expf(s2[1][g] - mnew);
            s2[0][g] = p0; s2[1][g] = p1;
            float ps = p0 + p1;
#pragma unroll
            for (int off = 1; off < 16; off <<= 1) ps += __shfl_xor(ps, off, 32);
            rsum[g] = rsum[g] * fg[g] + ps;
        }
#pragma unroll
        for (int nt = 0; nt < 8; ++nt)
#pragma unroll
            for (int g = 0; g < 8; ++g) accO[nt][g] *= fg[g];

        // ---- re-stripe P (C-layout) -> A-layout via per-wave LDS scratch ----
        __bf16* psc = &Ps[wave][0];
#pragma unroll
        for (int g = 0; g < 8; ++g) {
            int r = g + 8 * hi;
            psc[r * LDT + mm]      = f2bf(s2[0][g]);
            psc[r * LDT + 16 + mm] = f2bf(s2[1][g]);
        }
        v16bf pfr = load_frag(&psc[mm * LDT], 0, kh);   // same-wave LDS RAW: dscnt-ordered
        // ---- O += P(16x32) x V(32x128): 8 WMMAs ----
#pragma unroll
        for (int nt = 0; nt < 8; ++nt) {
            v16bf vfr = load_frag(&Vt[(nt * 16 + mm) * LDT], 0, kh);
            accO[nt] = __builtin_amdgcn_wmma_f32_16x16x32_bf16(
                false, pfr, false, vfr, (short)0, accO[nt], false, false);
        }
        __syncthreads();
    }

    // ---- epilogue: normalize + bf16 store ----
#pragma unroll
    for (int g = 0; g < 8; ++g) {
        float inv = 1.0f / rsum[g];
        int row = qB + g + 8 * hi;
#pragma unroll
        for (int nt = 0; nt < 8; ++nt)
            outb[(size_t)row * (NHEAD * VDIM) + h * VDIM + nt * 16 + mm] =
                f2bf(accO[nt][g] * inv);
    }
}

// ---------------- elementwise kernels ----------------
__global__ __launch_bounds__(256) void cvt_bf16_kernel(const float* __restrict__ in,
                                                       __bf16* __restrict__ o, size_t n) {
    size_t i = (size_t)blockIdx.x * blockDim.x + threadIdx.x;
    if (i < n) o[i] = f2bf(in[i]);
}

__global__ __launch_bounds__(256) void cvt_cols_kernel(const float* __restrict__ in,
                                                       __bf16* __restrict__ o,
                                                       int rows, int incols, int outcols) {
    size_t i = (size_t)blockIdx.x * blockDim.x + threadIdx.x;
    size_t n = (size_t)rows * outcols;
    if (i < n) {
        int r = (int)(i / outcols), c = (int)(i % outcols);
        o[i] = f2bf(in[(size_t)r * incols + c]);
    }
}

__global__ __launch_bounds__(256) void rms_bf16_kernel(const float* __restrict__ in,
                                                       const float* __restrict__ w,
                                                       __bf16* __restrict__ o, int C) {
    int row = blockIdx.x;
    const float* x = in + (size_t)row * C;
    float ss = 0.0f;
    for (int i = threadIdx.x; i < C; i += 256) { float v = x[i]; ss += v * v; }
    ss = block_sum_256(ss);
    float inv = rsqrtf(ss / (float)C + 1e-6f);
    for (int i = threadIdx.x; i < C; i += 256) o[(size_t)row * C + i] = f2bf(x[i] * inv * w[i]);
}

__global__ __launch_bounds__(256) void rms_add_kernel(const float* __restrict__ in,
                                                      const float* __restrict__ w,
                                                      const float* __restrict__ res,
                                                      float* __restrict__ o, int C) {
    int row = blockIdx.x;
    const float* x = in + (size_t)row * C;
    float ss = 0.0f;
    for (int i = threadIdx.x; i < C; i += 256) { float v = x[i]; ss += v * v; }
    ss = block_sum_256(ss);
    float inv = rsqrtf(ss / (float)C + 1e-6f);
    for (int i = threadIdx.x; i < C; i += 256)
        o[(size_t)row * C + i] = res[(size_t)row * C + i] + x[i] * inv * w[i];
}

__global__ __launch_bounds__(256) void ln_bf16_kernel(const float* __restrict__ in, int inStride,
                                                      int C, const float* __restrict__ sc,
                                                      const float* __restrict__ bi,
                                                      __bf16* __restrict__ o, int outStride) {
    int row = blockIdx.x;
    const float* x = in + (size_t)row * inStride;
    float sm = 0.0f, sq = 0.0f;
    for (int i = threadIdx.x; i < C; i += 256) { float v = x[i]; sm += v; sq += v * v; }
    sm = block_sum_256(sm);
    sq = block_sum_256(sq);
    float mean = sm / (float)C;
    float var  = sq / (float)C - mean * mean;
    float inv  = rsqrtf(var + 1e-6f);
    for (int i = threadIdx.x; i < C; i += 256)
        o[(size_t)row * outStride + i] = f2bf((x[i] - mean) * inv * sc[i] + bi[i]);
}

__global__ __launch_bounds__(256) void build_q_kernel(const float* __restrict__ q,
                                                      const int* __restrict__ pos,
                                                      __bf16* __restrict__ qf, int S) {
    size_t idx = (size_t)blockIdx.x * blockDim.x + threadIdx.x;
    size_t n = (size_t)S * NHEAD * QHD;
    if (idx >= n) return;
    int s = (int)(idx / (NHEAD * QHD));
    int t = (int)(idx % (NHEAD * QHD));
    int head = t / QHD, d = t % QHD;
    const float* qr = q + (size_t)s * (NHEAD * QHD) + head * QHD;
    float val;
    if (d < NOPE_D) {
        val = qr[d];
    } else {
        int r = d - NOPE_D;
        int i = r & 31;
        float inv = __powf(10000.0f, -(float)(2 * i) / 64.0f);
        float f = (float)pos[s] * inv;
        float x = qr[NOPE_D + r];
        float rot = (r < 32) ? -qr[NOPE_D + r + 32] : qr[NOPE_D + r - 32];
        val = x * __cosf(f) + rot * __sinf(f);
    }
    qf[idx] = f2bf(val);
}

__global__ __launch_bounds__(256) void build_kv_kernel(const float* __restrict__ ckv,
                                                       const float* __restrict__ kvs,
                                                       const int* __restrict__ pos,
                                                       __bf16* __restrict__ kf,
                                                       __bf16* __restrict__ v, int S) {
    size_t idx = (size_t)blockIdx.x * blockDim.x + threadIdx.x;
    size_t n = (size_t)S * 320;   // 192 (k) + 128 (v)
    if (idx >= n) return;
    int s = (int)(idx / 320);
    int d = (int)(idx % 320);
    if (d < QHD) {
        float val;
        if (d < NOPE_D) {
            val = kvs[(size_t)s * 256 + d];
        } else {
            int r = d - NOPE_D;
            int i = r & 31;
            float inv = __powf(10000.0f, -(float)(2 * i) / 64.0f);
            float f = (float)pos[s] * inv;
            float x = ckv[(size_t)s * 576 + KVLAT + r];
            float rot = (r < 32) ? -ckv[(size_t)s * 576 + KVLAT + r + 32]
                                 :  ckv[(size_t)s * 576 + KVLAT + r - 32];
            val = x * __cosf(f) + rot * __sinf(f);
        }
        kf[(size_t)s * QHD + d] = f2bf(val);
    } else {
        int dv = d - QHD;
        v[(size_t)s * VDIM + dv] = f2bf(kvs[(size_t)s * 256 + NOPE_D + dv]);
    }
}

__global__ __launch_bounds__(256) void silu_mul_kernel(const float* __restrict__ gu,
                                                       __bf16* __restrict__ o, int S) {
    size_t idx = (size_t)blockIdx.x * blockDim.x + threadIdx.x;
    size_t n = (size_t)S * INTERD;
    if (idx >= n) return;
    int s = (int)(idx / INTERD);
    int i = (int)(idx % INTERD);
    float g = gu[(size_t)s * (2 * INTERD) + i];
    float u = gu[(size_t)s * (2 * INTERD) + INTERD + i];
    float sig = 1.0f / (1.0f + __expf(-g));
    o[idx] = f2bf(u * g * sig);
}

// ---------------- host ----------------
extern "C" void kernel_launch(void* const* d_in, const int* in_sizes, int n_in,
                              void* d_out, int out_size, void* d_ws, size_t ws_size,
                              hipStream_t stream) {
    (void)in_sizes; (void)n_in; (void)out_size; (void)ws_size;
    const float* hidden    = (const float*)d_in[0];
    const int*   pos       = (const int*)d_in[2];
    const float* w_in      = (const float*)d_in[3];
    const float* w_pattn   = (const float*)d_in[4];
    const float* w_preff   = (const float*)d_in[5];
    const float* w_pff     = (const float*)d_in[6];
    const float* qa_w      = (const float*)d_in[7];
    const float* qa_ln_s   = (const float*)d_in[8];
    const float* qa_ln_b   = (const float*)d_in[9];
    const float* qb_w      = (const float*)d_in[10];
    const float* kv_mqa_w  = (const float*)d_in[11];
    const float* kv_ln_s   = (const float*)d_in[12];
    const float* kv_ln_b   = (const float*)d_in[13];
    const float* kvi_w     = (const float*)d_in[14];
    const float* o_w       = (const float*)d_in[15];
    const float* gu_w      = (const float*)d_in[16];
    const float* dn_w      = (const float*)d_in[17];
    float* out = (float*)d_out;

    char* base = (char*)d_ws;
    size_t off = 0;
    auto alloc = [&](size_t bytes) -> void* {
        size_t o = (off + 255) & ~(size_t)255;
        off = o + bytes;
        return (void*)(base + o);
    };
    const int S = S_LEN;
    __bf16* hb      = (__bf16*)alloc((size_t)S * HIDDIM * 2);
    __bf16* qa_wb   = (__bf16*)alloc((size_t)HIDDIM * QLAT * 2);
    __bf16* qb_wb   = (__bf16*)alloc((size_t)QLAT * NHEAD * QHD * 2);
    __bf16* kvm_wb  = (__bf16*)alloc((size_t)HIDDIM * 576 * 2);
    __bf16* kvi_wb  = (__bf16*)alloc((size_t)KVLAT * 256 * 2);
    __bf16* o_wb    = (__bf16*)alloc((size_t)HIDDIM * HIDDIM * 2);
    __bf16* gu_wb   = (__bf16*)alloc((size_t)HIDDIM * 2 * INTERD * 2);
    __bf16* dn_wb   = (__bf16*)alloc((size_t)INTERD * HIDDIM * 2);
    float*  qa      = (float*) alloc((size_t)S * QLAT * 4);
    __bf16* qa_lnb  = (__bf16*)alloc((size_t)S * QLAT * 2);
    float*  qfull   = (float*) alloc((size_t)S * NHEAD * QHD * 4);
    float*  ckv     = (float*) alloc((size_t)S * 576 * 4);
    __bf16* kv_lnb  = (__bf16*)alloc((size_t)S * KVLAT * 2);
    float*  kvs     = (float*) alloc((size_t)S * 256 * 4);
    __bf16* qfb     = (__bf16*)alloc((size_t)S * NHEAD * QHD * 2);
    __bf16* kfb     = (__bf16*)alloc((size_t)S * QHD * 2);
    __bf16* vb      = (__bf16*)alloc((size_t)S * VDIM * 2);
    __bf16* attn_ob = (__bf16*)alloc((size_t)S * HIDDIM * 2);
    float*  attn_p  = (float*) alloc((size_t)S * HIDDIM * 4);
    float*  xbuf    = (float*) alloc((size_t)S * HIDDIM * 4);
    __bf16* hffb    = (__bf16*)alloc((size_t)S * HIDDIM * 2);
    float*  gu      = (float*) alloc((size_t)S * 2 * INTERD * 4);
    __bf16* interb  = (__bf16*)alloc((size_t)S * INTERD * 2);
    float*  dnout   = (float*) alloc((size_t)S * HIDDIM * 4);

    auto cvt = [&](const float* src, __bf16* dst, size_t n) {
        cvt_bf16_kernel<<<(unsigned)((n + 255) / 256), 256, 0, stream>>>(src, dst, n);
    };
    auto gemm = [&](const __bf16* A, const __bf16* B, float* C, int M, int N, int K) {
        dim3 grid((N + BN - 1) / BN, M / BM);
        if (N % BN == 0)
            gemm_bf16_wmma<false><<<grid, 128, 0, stream>>>(A, B, C, M, N, K);
        else
            gemm_bf16_wmma<true><<<grid, 128, 0, stream>>>(A, B, C, M, N, K);
    };

    // weight conversion to bf16 (done every call; deterministic)
    cvt(qa_w,  qa_wb,  (size_t)HIDDIM * QLAT);
    cvt(qb_w,  qb_wb,  (size_t)QLAT * NHEAD * QHD);
    cvt(kv_mqa_w, kvm_wb, (size_t)HIDDIM * 576);
    cvt_cols_kernel<<<(unsigned)(((size_t)KVLAT * 256 + 255) / 256), 256, 0, stream>>>(
        kvi_w, kvi_wb, KVLAT, NHEAD * (NOPE_D + VDIM), 256);
    cvt(o_w,   o_wb,   (size_t)HIDDIM * HIDDIM);
    cvt(gu_w,  gu_wb,  (size_t)HIDDIM * 2 * INTERD);
    cvt(dn_w,  dn_wb,  (size_t)INTERD * HIDDIM);

    // h = rms(hidden, w_in) -> bf16
    rms_bf16_kernel<<<S, 256, 0, stream>>>(hidden, w_in, hb, HIDDIM);

    // q path
    gemm(hb, qa_wb, qa, S, QLAT, HIDDIM);
    ln_bf16_kernel<<<S, 256, 0, stream>>>(qa, QLAT, QLAT, qa_ln_s, qa_ln_b, qa_lnb, QLAT);
    gemm(qa_lnb, qb_wb, qfull, S, NHEAD * QHD, QLAT);

    // kv path
    gemm(hb, kvm_wb, ckv, S, 576, HIDDIM);
    ln_bf16_kernel<<<S, 256, 0, stream>>>(ckv, 576, KVLAT, kv_ln_s, kv_ln_b, kv_lnb, KVLAT);
    gemm(kv_lnb, kvi_wb, kvs, S, 256, KVLAT);

    // RoPE + bf16 packing
    build_q_kernel<<<(unsigned)(((size_t)S * NHEAD * QHD + 255) / 256), 256, 0, stream>>>(
        qfull, pos, qfb, S);
    build_kv_kernel<<<(unsigned)(((size_t)S * 320 + 255) / 256), 256, 0, stream>>>(
        ckv, kvs, pos, kfb, vb, S);

    // flash attention
    flash_attn_wmma<<<dim3(S / 64, NHEAD), 128, 0, stream>>>(qfb, kfb, vb, attn_ob, S);

    // output projection + residual
    gemm(attn_ob, o_wb, attn_p, S, HIDDIM, HIDDIM);
    rms_add_kernel<<<S, 256, 0, stream>>>(attn_p, w_pattn, hidden, xbuf, HIDDIM);

    // MLP
    rms_bf16_kernel<<<S, 256, 0, stream>>>(xbuf, w_preff, hffb, HIDDIM);
    gemm(hffb, gu_wb, gu, S, 2 * INTERD, HIDDIM);
    silu_mul_kernel<<<(unsigned)(((size_t)S * INTERD + 255) / 256), 256, 0, stream>>>(
        gu, interb, S);
    gemm(interb, dn_wb, dnout, S, HIDDIM, INTERD);
    rms_add_kernel<<<S, 256, 0, stream>>>(dnout, w_pff, xbuf, out, HIDDIM);
}
